// MockMobGatedDeltaNetMoE_57698590655107
// MI455X (gfx1250) — matrix operand order repaired and checked by reading, not compiled, not run on hardware
//
#include <hip/hip_runtime.h>
#include <hip/hip_bf16.h>

// ---------------------------------------------------------------------------
// Types for WMMA bf16 (gfx1250): V_WMMA_F32_16X16X32_BF16
// ---------------------------------------------------------------------------
typedef __attribute__((ext_vector_type(16))) __bf16     bf16x16;
typedef __attribute__((ext_vector_type(8)))  float      f32x8;
typedef __attribute__((ext_vector_type(8)))  unsigned   u32x8;

static __device__ __forceinline__ f32x8 wmma_bf16(bf16x16 a, bf16x16 b, f32x8 c) {
  // (neg_a, A, neg_b, B, c_mod, C, reuse_a, reuse_b)
  return __builtin_amdgcn_wmma_f32_16x16x32_bf16(false, a, false, b, (short)0, c,
                                                 false, false);
}

// Build a 16-element bf16 fragment from two 8-element (16B) chunks.
// A-fragment: chunks at K and K+16 (ISA A layout).  B-fragment: the
// same two chunks (K, K+16) as A.
static __device__ __forceinline__ bf16x16 frag2(const __bf16* lo, const __bf16* hi) {
  uint4 x = *reinterpret_cast<const uint4*>(lo);
  uint4 y = *reinterpret_cast<const uint4*>(hi);
  u32x8 t = {x.x, x.y, x.z, x.w, y.x, y.y, y.z, y.w};
  return __builtin_bit_cast(bf16x16, t);
}

static __device__ __forceinline__ uint4 ld16(const __bf16* p) {
  return *reinterpret_cast<const uint4*>(p);
}
static __device__ __forceinline__ void st16(__bf16* p, uint4 v) {
  *reinterpret_cast<uint4*>(p) = v;
}

// ---------------------------------------------------------------------------
// Problem constants
// ---------------------------------------------------------------------------
#define NH   8
#define HD   256
#define NR   6
#define NS   2
#define HIDN 2048
#define DVV  512      // HD*EV
#define KEYD 2048     // NH*HD
#define VALD 4096     // KEYD*EV
#define BB   2
#define TT   1024
#define BTOK 2048     // BB*TT
#define EXPN 1536     // NR*HD

// ---------------------------------------------------------------------------
// Elementwise f32 -> bf16 convert
// ---------------------------------------------------------------------------
__global__ void k_convert(const float* __restrict__ in, __bf16* __restrict__ out,
                          size_t n) {
  size_t i = (size_t)blockIdx.x * blockDim.x + threadIdx.x;
  if (i < n) out[i] = (__bf16)in[i];
}

// Transpose [rows, cols] f32 -> [cols, rows] bf16, batched over grid.y.
__global__ void k_transpose(const float* __restrict__ in, __bf16* __restrict__ out,
                            int rows, int cols) {
  size_t total = (size_t)rows * cols;
  size_t i = (size_t)blockIdx.x * blockDim.x + threadIdx.x;
  if (i >= total) return;
  size_t z  = blockIdx.y;
  int    c  = (int)(i / rows);
  int    rr = (int)(i % rows);
  out[z * total + i] = (__bf16)in[z * total + (size_t)rr * cols + c];
}

// v_s [bt, H*DV] bf16  ->  VT [(b*H+h)*DV + dv][t] bf16 (contiguous in t)
__global__ void k_transpose_v(const __bf16* __restrict__ v_s,
                              __bf16* __restrict__ VT) {
  size_t i = (size_t)blockIdx.x * blockDim.x + threadIdx.x;
  size_t total = (size_t)BB * NH * DVV * TT;
  if (i >= total) return;
  int t    = (int)(i & (TT - 1));
  size_t r = i >> 10;
  int dv   = (int)(r & (DVV - 1)); r >>= 9;
  int h    = (int)(r & (NH - 1));
  int b    = (int)(r >> 3);
  VT[i] = v_s[((size_t)b * TT + t) * VALD + h * DVV + dv];
}

// ---------------------------------------------------------------------------
// Routing: per (b,t,h): logits = q . Wgate, softmax(4), top2, scatter weights
// rw layout: [b, h, t, R] fp32
// ---------------------------------------------------------------------------
__global__ void k_routing(const __bf16* __restrict__ q_s,
                          const float* __restrict__ Wgate,
                          float* __restrict__ rw) {
  int gid = blockIdx.x * blockDim.x + threadIdx.x;
  if (gid >= BTOK * NH) return;
  int h  = gid & (NH - 1);
  int bt = gid >> 3;
  const __bf16* qp = q_s + (size_t)bt * KEYD + h * HD;
  float lg[4] = {0.f, 0.f, 0.f, 0.f};
  for (int d = 0; d < HD; ++d) {
    float qf = (float)qp[d];
#pragma unroll
    for (int n = 0; n < 4; ++n) lg[n] += qf * Wgate[d * 4 + n];
  }
  float mx = fmaxf(fmaxf(lg[0], lg[1]), fmaxf(lg[2], lg[3]));
  float e[4];
#pragma unroll
  for (int n = 0; n < 4; ++n) e[n] = __expf(lg[n] - mx);
  int i1 = 0;
#pragma unroll
  for (int n = 1; n < 4; ++n) if (e[n] > e[i1]) i1 = n;
  int i2 = -1;
#pragma unroll
  for (int n = 0; n < 4; ++n)
    if (n != i1 && (i2 < 0 || e[n] > e[i2])) i2 = n;
  float tw = e[i1] + e[i2];
  float out[NR] = {0.25f, 0.25f, 0.f, 0.f, 0.f, 0.f};
  out[NS + i1] = 0.5f * e[i1] / tw;
  out[NS + i2] = 0.5f * e[i2] / tw;
  int b = bt >> 10, t = bt & (TT - 1);
  float* dst = rw + (((size_t)b * NH + h) * TT + t) * NR;
#pragma unroll
  for (int r = 0; r < NR; ++r) dst[r] = out[r];
}

// ---------------------------------------------------------------------------
// Generic bf16 WMMA GEMM:  C[M,N] = A[M,K] * Bt[N,K]^T
// Workgroup tile 128x128, 8 waves (4x2), each wave 32x64 (2x4 wmma tiles).
// Register-staged double-buffered LDS: global loads for step k+1 issue before
// the WMMAs of step k; their waitcnt lands at the ds_store into the alternate
// buffer.  One barrier per 32-wide K step.
// MODE 0: store bf16     MODE 1: store bf16 * (1/16)   (qe, folds 1/sqrt(D))
// MODE 2: store bf16 * key-mask(rw)   (ke)             MODE 3: store fp32
// M,N multiples of 128 and K multiple of 32 for this problem.
// ---------------------------------------------------------------------------
template <int MODE>
__global__ __launch_bounds__(256) void k_gemm(
    const __bf16* __restrict__ A, long lda, long aZ,
    const __bf16* __restrict__ Bt, long ldbt, long bZ,
    void* __restrict__ Cp, long ldc, long cZ,
    int K, const float* __restrict__ rw) {
  __shared__ __bf16 sA[2][128 * 32];
  __shared__ __bf16 sB[2][128 * 32];
  const int tid = threadIdx.x, lane = tid & 31, w = tid >> 5;
  const int wm = w & 3, wn = w >> 2;
  const int z = blockIdx.z;
  const int row0 = blockIdx.y * 128, col0 = blockIdx.x * 128;

  // Loader coords: 256 threads cover 128 rows x 32 cols (16 bf16 each).
  const int rl = tid >> 1, cl = (tid & 1) * 16;
  const __bf16* aRow = A + (size_t)z * aZ + (size_t)(row0 + rl) * lda;
  const __bf16* bRow = Bt + (size_t)z * bZ + (size_t)(col0 + rl) * ldbt;

  uint4 ra0 = ld16(aRow + cl), ra1 = ld16(aRow + cl + 8);
  uint4 rb0 = ld16(bRow + cl), rb1 = ld16(bRow + cl + 8);
  st16(&sA[0][rl * 32 + cl], ra0); st16(&sA[0][rl * 32 + cl + 8], ra1);
  st16(&sB[0][rl * 32 + cl], rb0); st16(&sB[0][rl * 32 + cl + 8], rb1);
  __syncthreads();

  f32x8 acc[2][4] = {};
  const int ln = lane & 15, kb = (lane >> 4) * 8, kb2 = (lane >> 4) * 16;

  for (int kk = 0; kk < K; kk += 32) {
    const int cur = (kk >> 5) & 1;
    const bool more = (kk + 32) < K;
    if (more) {  // issue next-stage global loads before this stage's math
      ra0 = ld16(aRow + kk + 32 + cl); ra1 = ld16(aRow + kk + 32 + cl + 8);
      rb0 = ld16(bRow + kk + 32 + cl); rb1 = ld16(bRow + kk + 32 + cl + 8);
    }
    bf16x16 af[2];
#pragma unroll
    for (int rt = 0; rt < 2; ++rt) {
      const int m = wm * 32 + rt * 16 + ln;
      af[rt] = frag2(&sA[cur][m * 32 + kb], &sA[cur][m * 32 + 16 + kb]);
    }
#pragma unroll
    for (int f = 0; f < 4; ++f) {
      const int n = wn * 64 + f * 16 + ln;
      bf16x16 bfr = frag2(&sB[cur][n * 32 + kb], &sB[cur][n * 32 + 16 + kb]);
      acc[0][f] = wmma_bf16(af[0], bfr, acc[0][f]);
      acc[1][f] = wmma_bf16(af[1], bfr, acc[1][f]);
    }
    if (more) {
      const int nxt = cur ^ 1;
      st16(&sA[nxt][rl * 32 + cl], ra0); st16(&sA[nxt][rl * 32 + cl + 8], ra1);
      st16(&sB[nxt][rl * 32 + cl], rb0); st16(&sB[nxt][rl * 32 + cl + 8], rb1);
    }
    __syncthreads();
  }

#pragma unroll
  for (int rt = 0; rt < 2; ++rt) {
#pragma unroll
    for (int f = 0; f < 4; ++f) {
#pragma unroll
      for (int i = 0; i < 8; ++i) {
        const int row = row0 + wm * 32 + rt * 16 + (lane >> 4) * 8 + i;
        const int col = col0 + wn * 64 + f * 16 + ln;
        float v = acc[rt][f][i];
        if (MODE == 1) v *= 0.0625f;  // 1/sqrt(256)
        if (MODE == 2) {
          const int b = row >> 10, t = row & (TT - 1), r = col >> 8;
          if (rw[(((size_t)b * NH + z) * TT + t) * NR + r] == 0.0f) v = 0.0f;
        }
        if (MODE == 3)
          ((float*)Cp)[(size_t)z * cZ + (size_t)row * ldc + col] = v;
        else
          ((__bf16*)Cp)[(size_t)z * cZ + (size_t)row * ldc + col] = (__bf16)v;
      }
    }
  }
}

// ---------------------------------------------------------------------------
// Attention: one workgroup = (b, h, 32-query tile).  Loops experts r=0..5,
// accumulating P_total[m,k] = sum_r rw_r[m]*mask_r[k]*softmax(qe_r ke_r^T)[m,k]
// in LDS, then a single WMMA pass O = P_total @ V, fused silu(g) gate.
// LDS: S 128KB + Pacc 128KB + reductions (< 320KB WGP budget).
// ---------------------------------------------------------------------------
__global__ __launch_bounds__(256) void k_attn(
    const __bf16* __restrict__ qe,  // [h][bt][R*HD], pre-scaled by 1/16
    const __bf16* __restrict__ ke,  // [h][bt][R*HD], key-masked
    const __bf16* __restrict__ VT,  // [(b*H+h)*DV + dv][t]
    const float* __restrict__ rw,   // [b][h][t][R]
    const __bf16* __restrict__ g_s, // [bt][H*DV]
    __bf16* __restrict__ comb) {    // [bt][H*DV]
  extern __shared__ float sm[];
  float* S        = sm;                   // 32*1024
  float* Pacc     = sm + 32 * 1024;       // 32*1024
  float* red      = Pacc + 32 * 1024;     // 256
  float* rowmax   = red + 256;            // 32
  float* rowsum   = rowmax + 32;          // 32
  float* rowscale = rowsum + 32;          // 32
  float* kmask    = rowscale + 32;        // 1024
  __bf16* Pbf = (__bf16*)S;               // aliases S after last use

  const int tid = threadIdx.x, lane = tid & 31, w = tid >> 5;
  const int q0 = blockIdx.x * 32, h = blockIdx.y, b = blockIdx.z;
  const size_t bhT = ((size_t)b * NH + h) * TT;

  for (int e = tid; e < 32 * 1024; e += 256) Pacc[e] = 0.f;

  const size_t qeBase = ((size_t)h * BTOK + b * TT + q0) * EXPN;
  const size_t keBase = ((size_t)h * BTOK + b * TT) * EXPN;

  for (int r = 0; r < NR; ++r) {
    // ---- S = qe * ke^T (wave w owns key columns [w*128, w*128+128)) ----
    const __bf16* qeP = qe + qeBase + r * HD;
    const __bf16* keP = ke + keBase + r * HD;
    const __bf16* pa  = qeP + (size_t)(lane & 15) * EXPN + ((lane >> 4) * 8);
#pragma unroll 1
    for (int ct = 0; ct < 8; ++ct) {
      const int col0 = w * 128 + ct * 16;
      const int key  = col0 + (lane & 15);
      const __bf16* pb = keP + (size_t)key * EXPN + ((lane >> 4) * 8);
      f32x8 a0 = {}, a1 = {};
#pragma unroll
      for (int ks = 0; ks < 8; ++ks) {
        bf16x16 A0 = frag2(pa + ks * 32, pa + ks * 32 + 16);
        bf16x16 A1 = frag2(pa + 16 * EXPN + ks * 32, pa + 16 * EXPN + ks * 32 + 16);
        bf16x16 B0 = frag2(pb + ks * 32, pb + ks * 32 + 16);
        a0 = wmma_bf16(A0, B0, a0);
        a1 = wmma_bf16(A1, B0, a1);
      }
#pragma unroll
      for (int i = 0; i < 8; ++i) {
        const int rr = (lane >> 4) * 8 + i;
        S[rr * 1024 + col0 + (lane & 15)]        = a0[i];
        S[(16 + rr) * 1024 + col0 + (lane & 15)] = a1[i];
      }
    }
    __syncthreads();

    // ---- row stats (8 threads per row) + key masks + row weights ----
    const int row = tid >> 3, c0 = (tid & 7) * 128;
    float pm = -3.0e38f;
    for (int j = 0; j < 128; ++j) pm = fmaxf(pm, S[row * 1024 + c0 + j]);
    red[tid] = pm;
#pragma unroll
    for (int u = 0; u < 4; ++u) {
      const int key = tid * 4 + u;
      kmask[key] = (rw[(bhT + key) * NR + r] != 0.f) ? 1.f : 0.f;
    }
    if (tid < 32) rowscale[tid] = rw[(bhT + q0 + tid) * NR + r];
    __syncthreads();
    if (tid < 32) {
      float m2 = red[tid * 8];
#pragma unroll
      for (int u = 1; u < 8; ++u) m2 = fmaxf(m2, red[tid * 8 + u]);
      rowmax[tid] = m2;
    }
    __syncthreads();
    const float rm = rowmax[row];
    float ps = 0.f;
    for (int j = 0; j < 128; ++j) ps += __expf(S[row * 1024 + c0 + j] - rm);
    red[tid] = ps;
    __syncthreads();
    if (tid < 32) {
      float s2 = 0.f;
#pragma unroll
      for (int u = 0; u < 8; ++u) s2 += red[tid * 8 + u];
      rowsum[tid] = s2;
    }
    __syncthreads();
    const float sc = rowscale[row] / rowsum[row];
    if (sc != 0.f) {
      for (int j = 0; j < 128; ++j) {
        const int c = c0 + j;
        Pacc[row * 1024 + c] += __expf(S[row * 1024 + c] - rm) * sc * kmask[c];
      }
    }
    __syncthreads();
  }

  // ---- P_total -> bf16 (overwrites S region) ----
  for (int e = tid; e < 32 * 1024; e += 256) Pbf[e] = (__bf16)Pacc[e];
  __syncthreads();

  // ---- O = P_total @ V  (wave w owns dv columns [w*64, w*64+64)) ----
  f32x8 acc[2][4] = {};
  const __bf16* VTb = VT + (((size_t)b * NH + h) * DVV) * TT;
  for (int kt = 0; kt < 32; ++kt) {
    const int m = lane & 15, kb = (lane >> 4) * 8;
    const __bf16* p0 = &Pbf[m * 1024 + kt * 32 + kb];
    const __bf16* p1 = &Pbf[(16 + m) * 1024 + kt * 32 + kb];
    bf16x16 A0 = frag2(p0, p0 + 16);
    bf16x16 A1 = frag2(p1, p1 + 16);
#pragma unroll
    for (int f = 0; f < 4; ++f) {
      const int dv = w * 64 + f * 16 + (lane & 15);
      const __bf16* pb = VTb + (size_t)dv * TT + kt * 32 + ((lane >> 4) * 8);
      bf16x16 B0 = frag2(pb, pb + 16);
      acc[0][f] = wmma_bf16(A0, B0, acc[0][f]);
      acc[1][f] = wmma_bf16(A1, B0, acc[1][f]);
    }
  }

  // ---- fused gate: comb = O * silu(g) ----
#pragma unroll
  for (int rt = 0; rt < 2; ++rt)
#pragma unroll
    for (int f = 0; f < 4; ++f)
#pragma unroll
      for (int i = 0; i < 8; ++i) {
        const int qrow = q0 + rt * 16 + (lane >> 4) * 8 + i;
        const int hdv  = h * DVV + w * 64 + f * 16 + (lane & 15);
        const size_t idx = ((size_t)b * TT + qrow) * VALD + hdv;
        const float gv  = (float)g_s[idx];
        const float sil = gv / (1.f + __expf(-gv));
        comb[idx] = (__bf16)(acc[rt][f][i] * sil);
      }
}

// ---------------------------------------------------------------------------
// Host-side orchestration
// ---------------------------------------------------------------------------
extern "C" void kernel_launch(void* const* d_in, const int* in_sizes, int n_in,
                              void* d_out, int out_size, void* d_ws, size_t ws_size,
                              hipStream_t stream) {
  const float* hs    = (const float*)d_in[0];
  const float* Wq    = (const float*)d_in[1];
  const float* Wk    = (const float*)d_in[2];
  const float* Wv    = (const float*)d_in[3];
  const float* Wqexp = (const float*)d_in[4];
  const float* Wkexp = (const float*)d_in[5];
  const float* Wgate = (const float*)d_in[6];
  const float* Wg    = (const float*)d_in[7];
  const float* Wo    = (const float*)d_in[8];
  float* out = (float*)d_out;

  size_t off = 0;
  auto alloc = [&](size_t bytes) {
    void* p = (char*)d_ws + off;
    off += (bytes + 255) & ~(size_t)255;
    return p;
  };
  __bf16* hs_bf   = (__bf16*)alloc((size_t)BTOK * HIDN * 2);
  __bf16* WqT     = (__bf16*)alloc((size_t)KEYD * HIDN * 2);
  __bf16* WkT     = (__bf16*)alloc((size_t)KEYD * HIDN * 2);
  __bf16* WvT     = (__bf16*)alloc((size_t)VALD * HIDN * 2);
  __bf16* WgT     = (__bf16*)alloc((size_t)VALD * HIDN * 2);
  __bf16* WoT     = (__bf16*)alloc((size_t)HIDN * VALD * 2);
  __bf16* WqeT    = (__bf16*)alloc((size_t)NH * EXPN * HD * 2);
  __bf16* WkeT    = (__bf16*)alloc((size_t)NH * EXPN * HD * 2);
  __bf16* q_s     = (__bf16*)alloc((size_t)BTOK * KEYD * 2);
  __bf16* k_s     = (__bf16*)alloc((size_t)BTOK * KEYD * 2);
  __bf16* v_s     = (__bf16*)alloc((size_t)BTOK * VALD * 2);
  __bf16* g_s     = (__bf16*)alloc((size_t)BTOK * VALD * 2);
  __bf16* VTs     = (__bf16*)alloc((size_t)BB * NH * DVV * TT * 2);
  float*  rw_s    = (float*)alloc((size_t)BB * NH * TT * NR * 4);
  __bf16* qe_s    = (__bf16*)alloc((size_t)NH * BTOK * EXPN * 2);
  __bf16* ke_s    = (__bf16*)alloc((size_t)NH * BTOK * EXPN * 2);
  __bf16* comb    = (__bf16*)alloc((size_t)BTOK * VALD * 2);
  (void)ws_size; (void)n_in; (void)in_sizes; (void)out_size;

  const int TPB = 256;
  // 1) conversions / transposes
  k_convert<<<(BTOK * HIDN + TPB - 1) / TPB, TPB, 0, stream>>>(hs, hs_bf,
                                                               (size_t)BTOK * HIDN);
  k_transpose<<<dim3((HIDN * KEYD + TPB - 1) / TPB, 1), TPB, 0, stream>>>(
      Wq, WqT, HIDN, KEYD);
  k_transpose<<<dim3((HIDN * KEYD + TPB - 1) / TPB, 1), TPB, 0, stream>>>(
      Wk, WkT, HIDN, KEYD);
  k_transpose<<<dim3((HIDN * VALD + TPB - 1) / TPB, 1), TPB, 0, stream>>>(
      Wv, WvT, HIDN, VALD);
  k_transpose<<<dim3((HIDN * VALD + TPB - 1) / TPB, 1), TPB, 0, stream>>>(
      Wg, WgT, HIDN, VALD);
  k_transpose<<<dim3((VALD * HIDN + TPB - 1) / TPB, 1), TPB, 0, stream>>>(
      Wo, WoT, VALD, HIDN);
  k_transpose<<<dim3((HD * EXPN + TPB - 1) / TPB, NH), TPB, 0, stream>>>(
      Wqexp, WqeT, HD, EXPN);
  k_transpose<<<dim3((HD * EXPN + TPB - 1) / TPB, NH), TPB, 0, stream>>>(
      Wkexp, WkeT, HD, EXPN);

  // 2) base projections q, k, v, g  (tiles are 128x128 now)
  k_gemm<0><<<dim3(KEYD / 128, BTOK / 128, 1), TPB, 0, stream>>>(
      hs_bf, HIDN, 0, WqT, HIDN, 0, q_s, KEYD, 0, HIDN, nullptr);
  k_gemm<0><<<dim3(KEYD / 128, BTOK / 128, 1), TPB, 0, stream>>>(
      hs_bf, HIDN, 0, WkT, HIDN, 0, k_s, KEYD, 0, HIDN, nullptr);
  k_gemm<0><<<dim3(VALD / 128, BTOK / 128, 1), TPB, 0, stream>>>(
      hs_bf, HIDN, 0, WvT, HIDN, 0, v_s, VALD, 0, HIDN, nullptr);
  k_gemm<0><<<dim3(VALD / 128, BTOK / 128, 1), TPB, 0, stream>>>(
      hs_bf, HIDN, 0, WgT, HIDN, 0, g_s, VALD, 0, HIDN, nullptr);

  // 3) routing
  k_routing<<<(BTOK * NH + TPB - 1) / TPB, TPB, 0, stream>>>(q_s, Wgate, rw_s);

  // 4) expert expansions (batched over heads): qe scaled by 1/16, ke masked
  k_gemm<1><<<dim3(EXPN / 128, BTOK / 128, NH), TPB, 0, stream>>>(
      q_s, KEYD, HD, WqeT, HD, (long)EXPN * HD, qe_s, EXPN, (long)BTOK * EXPN,
      HD, nullptr);
  k_gemm<2><<<dim3(EXPN / 128, BTOK / 128, NH), TPB, 0, stream>>>(
      k_s, KEYD, HD, WkeT, HD, (long)EXPN * HD, ke_s, EXPN, (long)BTOK * EXPN,
      HD, rw_s);

  // 5) V transpose for P@V B-fragments
  k_transpose_v<<<(int)(((size_t)BB * NH * DVV * TT + TPB - 1) / TPB), TPB, 0,
                  stream>>>(v_s, VTs);

  // 6) attention (+ fused silu gate) -> comb
  const size_t attn_lds = (size_t)(32 * 1024 * 2 + 256 + 32 * 3 + 1024) * 4;
  k_attn<<<dim3(TT / 32, NH, BB), TPB, attn_lds, stream>>>(qe_s, ke_s, VTs,
                                                           rw_s, g_s, comb);

  // 7) final projection -> fp32 output
  k_gemm<3><<<dim3(HIDN / 128, BTOK / 128, 1), TPB, 0, stream>>>(
      comb, VALD, 0, WoT, VALD, 0, out, HIDN, 0, VALD, nullptr);
}
